// MLBMatchupPredictor_89455578841338
// MI455X (gfx1250) — compile-verified, matrix operand order checked by reading, not run on hardware
//
#include <hip/hip_runtime.h>
#include <hip/hip_bf16.h>
#include <math.h>

#define E_EDGES 500000
#define NPITCH  2000
#define NBATT   10000

typedef __attribute__((ext_vector_type(16))) _Float16 v16h;
typedef __attribute__((ext_vector_type(8)))  _Float16 h8;
typedef __attribute__((ext_vector_type(8)))  float    v8f;
typedef __attribute__((ext_vector_type(4)))  float    f4;

// ---------------------------------------------------------------------------
// WMMA helpers (gfx1250: V_WMMA_F32_16X16X32_F16, f32 accumulate)
// ---------------------------------------------------------------------------
__device__ __forceinline__ v8f wmma16(v16h a, v16h b, v8f c) {
  // 8 args: (neg_a, A, neg_b, B, c_mod, C, reuse_a, reuse_b)
  return __builtin_amdgcn_wmma_f32_16x16x32_f16(false, a, false, b, (short)0, c,
                                                false, false);
}

// 16-bit A (16x32) per-lane layout = two contiguous 8-element K chunks:
//   lanes 0-15 : K = {0..7} (v0..v3) and {16..23} (v4..v7)
//   lanes 16-31: K = {8..15}         and {24..31}
// B uses the identical pattern when weights are stored transposed [N][Kpad].

// Full 16x32 A tile from f32 row-major memory (row supplied per-lane so the
// caller can gather). All K valid. 4x b128 loads, no predication.
__device__ __forceinline__ v16h load_a_f32_full(const float* __restrict__ base, int row,
                                                int lda, int k0) {
  int lane = threadIdx.x & 31;
  int kb = (lane >> 4) << 3;
  const float* rp = base + (long)row * lda + k0 + kb;
  f4 c0 = *(const f4*)(rp);
  f4 c1 = *(const f4*)(rp + 4);
  f4 c2 = *(const f4*)(rp + 16);
  f4 c3 = *(const f4*)(rp + 20);
  v16h a;
#pragma unroll
  for (int i = 0; i < 4; ++i) {
    a[i]      = (_Float16)c0[i];
    a[4 + i]  = (_Float16)c1[i];
    a[8 + i]  = (_Float16)c2[i];
    a[12 + i] = (_Float16)c3[i];
  }
  return a;
}

// 16x32 A tile where only K=0..15 exists in memory (zero-pad K=16..31).
// Chunk at kb is always fully valid; chunk at kb+16 is always padding.
__device__ __forceinline__ v16h load_a_f32_pad16(const float* __restrict__ base, int row,
                                                 int lda) {
  int lane = threadIdx.x & 31;
  int kb = (lane >> 4) << 3;
  const float* rp = base + (long)row * lda + kb;
  f4 c0 = *(const f4*)(rp);
  f4 c1 = *(const f4*)(rp + 4);
  v16h a;
#pragma unroll
  for (int i = 0; i < 4; ++i) {
    a[i]      = (_Float16)c0[i];
    a[4 + i]  = (_Float16)c1[i];
    a[8 + i]  = (_Float16)0.0f;
    a[12 + i] = (_Float16)0.0f;
  }
  return a;
}

// 16x32 tile from f16 row-major [rows][lda] storage (transposed weights as B,
// or A re-read from LDS). Two aligned 16-byte loads per lane.
template <typename PTR>
__device__ __forceinline__ v16h load_t_f16(PTR base, int row, int lda, int k0) {
  int lane = threadIdx.x & 31;
  int kb = (lane >> 4) << 3;
  auto* rp = base + (long)row * lda + k0 + kb;
  h8 lo = *(const h8*)(rp);
  h8 hi = *(const h8*)(rp + 16);
  v16h a;
#pragma unroll
  for (int i = 0; i < 8; ++i) {
    a[i] = lo[i];
    a[8 + i] = hi[i];
  }
  return a;
}

// Order-preserving float<->uint encoding for atomicMax on floats.
__device__ __forceinline__ unsigned fenc(float f) {
  unsigned u = __float_as_uint(f);
  return (u & 0x80000000u) ? ~u : (u | 0x80000000u);
}
__device__ __forceinline__ float fdec(unsigned e) {
  unsigned u = (e & 0x80000000u) ? (e ^ 0x80000000u) : ~e;
  return __uint_as_float(u);
}

// ---------------------------------------------------------------------------
// Prep: W[K][N] f32 -> Wt[N][Kpad] f16 (zero K-padding)
// ---------------------------------------------------------------------------
__global__ void transpose_h_kernel(_Float16* __restrict__ dst, const float* __restrict__ src,
                                   int K, int N, int Kpad) {
  int i = blockIdx.x * 256 + threadIdx.x;
  if (i >= N * Kpad) return;
  int n = i / Kpad, k = i % Kpad;
  dst[i] = (k < K) ? (_Float16)src[k * N + n] : (_Float16)0.0f;
}

__global__ void zero_kernel(float* __restrict__ p, long n) {
  long i = (long)blockIdx.x * 256 + threadIdx.x;
  if (i < n) p[i] = 0.0f;
}

// ---------------------------------------------------------------------------
// Node transform: out[M,128] = A[M,K] @ Wt^T + bias   (one wave per 16x16 tile)
// ---------------------------------------------------------------------------
__global__ __launch_bounds__(32) void gemm_bias_kernel(
    const float* __restrict__ A, const _Float16* __restrict__ Bt,
    const float* __restrict__ bias, float* __restrict__ out, int N, int K, int tilesN) {
  int lane = threadIdx.x & 31;
  int m0 = (blockIdx.x / tilesN) << 4;
  int n0 = (blockIdx.x % tilesN) << 4;
  v8f c = {};
  for (int k0 = 0; k0 < K; k0 += 32) {
    v16h a = load_a_f32_full(A, m0 + (lane & 15), K, k0);
    v16h b = load_t_f16(Bt, n0 + (lane & 15), K, k0);
    c = wmma16(a, b, c);
  }
  int n = n0 + (lane & 15);
  float bb = bias[n];
#pragma unroll
  for (int v = 0; v < 8; ++v) {
    int m = m0 + v + ((lane >> 4) << 3);
    out[(long)m * N + n] = c[v] + bb;
  }
}

// ---------------------------------------------------------------------------
// GAT edge pass 1: alpha[e,h] = lrelu(xl[src]+xr[dst]+ee) . att[h]
// ee tile computed with 8 WMMAs per 16 edges; segment max via atomicMax.
// ---------------------------------------------------------------------------
__global__ __launch_bounds__(64) void gat_edge_alpha_kernel(
    const float* __restrict__ eattr, const int* __restrict__ srcIdx,
    const int* __restrict__ dstIdx, const _Float16* __restrict__ WeT,
    const float* __restrict__ xl, const float* __restrict__ xr,
    const float* __restrict__ att, float* __restrict__ alpha, unsigned* __restrict__ amax) {
  int wave = threadIdx.x >> 5;
  int lane = threadIdx.x & 31;
  int tile = blockIdx.x * 2 + wave;
  int e0 = tile << 4;
  int sE = srcIdx[e0 + (lane & 15)];
  int dE = dstIdx[e0 + (lane & 15)];
  // A = edge_attr[16 edges][K=16 padded to 32]
  v16h a = load_a_f32_pad16(eattr, e0 + (lane & 15), 16);
  float p0[8], p1[8];
#pragma unroll
  for (int v = 0; v < 8; ++v) { p0[v] = 0.0f; p1[v] = 0.0f; }
#pragma unroll
  for (int t8 = 0; t8 < 8; ++t8) {
    v16h b = load_t_f16(WeT, (t8 << 4) + (lane & 15), 32, 0);
    v8f c = {};
    c = wmma16(a, b, c);
    int ch = (t8 << 4) + (lane & 15);
    float w = att[ch];  // att flat [2*64] == [ch]
#pragma unroll
    for (int v = 0; v < 8; ++v) {
      int el = v + ((lane >> 4) << 3);
      int s = __shfl(sE, el, 32);
      int d = __shfl(dE, el, 32);
      float val = c[v] + xl[(long)s * 128 + ch] + xr[(long)d * 128 + ch];
      val = (val > 0.0f) ? val : 0.2f * val;  // leaky_relu 0.2
      if (t8 < 4) p0[v] += val * w; else p1[v] += val * w;
    }
  }
  // butterfly reduce over each 16-lane half: every lane ends with all 8 sums
#pragma unroll
  for (int v = 0; v < 8; ++v) {
#pragma unroll
    for (int off = 1; off < 16; off <<= 1) {
      p0[v] += __shfl_xor(p0[v], off, 32);
      p1[v] += __shfl_xor(p1[v], off, 32);
    }
  }
  int vsel = lane & 7;
  int el = vsel + ((lane >> 4) << 3);
  int d = __shfl(dE, el, 32);  // uniform EXEC here
  float a0 = 0.0f, a1 = 0.0f;
#pragma unroll
  for (int i = 0; i < 8; ++i)
    if (i == vsel) { a0 = p0[i]; a1 = p1[i]; }
  if ((lane & 15) < 8) {
    long ge = e0 + el;
    alpha[ge * 2 + 0] = a0;
    alpha[ge * 2 + 1] = a1;
    atomicMax(&amax[d * 2 + 0], fenc(a0));
    atomicMax(&amax[d * 2 + 1], fenc(a1));
  }
}

// ---------------------------------------------------------------------------
// GAT edge pass 2: ex = exp(alpha - amax[dst]); acc[dst] += ex*xl[src];
// denom[dst] += ex. 32 threads (4 channels each) per edge.
// ---------------------------------------------------------------------------
__global__ __launch_bounds__(256) void gat_edge_accum_kernel(
    const int* __restrict__ srcIdx, const int* __restrict__ dstIdx,
    const float* __restrict__ alpha, const unsigned* __restrict__ amax,
    const float* __restrict__ xl, float* __restrict__ denom, float* __restrict__ acc) {
  long gid = (long)blockIdx.x * 256 + threadIdx.x;
  int e = (int)(gid >> 5);
  int g = (int)(gid & 31);
  int s = srcIdx[e], d = dstIdx[e];
  float ex0 = expf(alpha[(long)e * 2 + 0] - fdec(amax[d * 2 + 0]));
  float ex1 = expf(alpha[(long)e * 2 + 1] - fdec(amax[d * 2 + 1]));
  if (g == 0) {
    atomicAdd(&denom[d * 2 + 0], ex0);
    atomicAdd(&denom[d * 2 + 1], ex1);
  }
  int ch0 = g << 2;
#pragma unroll
  for (int i = 0; i < 4; ++i) {
    int ch = ch0 + i;
    float ex = (ch < 64) ? ex0 : ex1;
    atomicAdd(&acc[(long)d * 128 + ch], ex * xl[(long)s * 128 + ch]);
  }
}

__global__ void gat_finalize_concat_kernel(const float* __restrict__ acc,
                                           const float* __restrict__ denom,
                                           const float* __restrict__ bias,
                                           float* __restrict__ out, int total) {
  int i = blockIdx.x * 256 + threadIdx.x;
  if (i >= total) return;
  int n = i >> 7, ch = i & 127, h = ch >> 6;
  float v = acc[i] / (denom[n * 2 + h] + 1e-16f) + bias[ch];
  out[i] = fmaxf(v, 0.0f);
}

__global__ void gat_finalize_mean_kernel(const float* __restrict__ acc,
                                         const float* __restrict__ denom,
                                         const float* __restrict__ bias,
                                         float* __restrict__ out, int total) {
  int i = blockIdx.x * 256 + threadIdx.x;
  if (i >= total) return;
  int n = i >> 6, c = i & 63;
  float v0 = acc[(long)n * 128 + c] / (denom[n * 2 + 0] + 1e-16f);
  float v1 = acc[(long)n * 128 + 64 + c] / (denom[n * 2 + 1] + 1e-16f);
  float v = 0.5f * (v0 + v1) + bias[c];
  out[i] = fmaxf(v, 0.0f);
}

// ---------------------------------------------------------------------------
// Fused edge classifier: q/k/e GEMMs -> sigmoid gate -> combine -> o1 GEMM
// -> o2 logits. One wave per 16 edges, 28 WMMAs/wave, zero HBM intermediates.
// ---------------------------------------------------------------------------
__global__ __launch_bounds__(64) void classifier_kernel(
    const int* __restrict__ srcIdx, const int* __restrict__ dstIdx,
    const float* __restrict__ eattr, const float* __restrict__ pitcher2,
    const float* __restrict__ batter2, const _Float16* __restrict__ qWT,
    const float* __restrict__ q_b, const _Float16* __restrict__ kWT,
    const float* __restrict__ k_b, const _Float16* __restrict__ eWT,
    const float* __restrict__ e_b, const _Float16* __restrict__ o1WT,
    const float* __restrict__ o1_b, const float* __restrict__ o2W,
    const float* __restrict__ o2_b, float* __restrict__ out) {
  __shared__ _Float16 lds[2][16 * 64];
  int wave = threadIdx.x >> 5;
  int lane = threadIdx.x & 31;
  int tile = blockIdx.x * 2 + wave;
  int e0 = tile << 4;
  int nn = lane & 15;
  int srow = srcIdx[e0 + nn];
  int drow = dstIdx[e0 + nn];

  v16h aq0 = load_a_f32_full(pitcher2, srow, 64, 0);
  v16h aq1 = load_a_f32_full(pitcher2, srow, 64, 32);
  v16h ak0 = load_a_f32_full(batter2, drow, 64, 0);
  v16h ak1 = load_a_f32_full(batter2, drow, 64, 32);
  v16h ae  = load_a_f32_pad16(eattr, e0 + nn, 16);

  v8f qd[4], kd[4], ed[4];
#pragma unroll
  for (int nt = 0; nt < 4; ++nt) {
    int nb = (nt << 4) + nn;
    v8f c = {};
    c = wmma16(aq0, load_t_f16(qWT, nb, 64, 0), c);
    c = wmma16(aq1, load_t_f16(qWT, nb, 64, 32), c);
    float qb = q_b[nb];
#pragma unroll
    for (int v = 0; v < 8; ++v) c[v] += qb;
    qd[nt] = c;
    v8f c2 = {};
    c2 = wmma16(ak0, load_t_f16(kWT, nb, 64, 0), c2);
    c2 = wmma16(ak1, load_t_f16(kWT, nb, 64, 32), c2);
    float kb = k_b[nb];
#pragma unroll
    for (int v = 0; v < 8; ++v) c2[v] += kb;
    kd[nt] = c2;
    v8f c3 = {};
    c3 = wmma16(ae, load_t_f16(eWT, nb, 32, 0), c3);
    float eb = e_b[nb];
#pragma unroll
    for (int v = 0; v < 8; ++v) c3[v] += eb;
    ed[nt] = c3;
  }

  // per-edge attention gate: aw = sigmoid(sum_c q*k)
  float s[8];
#pragma unroll
  for (int v = 0; v < 8; ++v) {
    s[v] = qd[0][v] * kd[0][v] + qd[1][v] * kd[1][v] +
           qd[2][v] * kd[2][v] + qd[3][v] * kd[3][v];
#pragma unroll
    for (int off = 1; off < 16; off <<= 1) s[v] += __shfl_xor(s[v], off, 32);
  }
  float aw[8];
#pragma unroll
  for (int v = 0; v < 8; ++v) aw[v] = 1.0f / (1.0f + expf(-s[v]));

  // combined = aw*k + (1-aw)*q + e  -> LDS (f16) for D->A relayout
  _Float16* my = lds[wave];
#pragma unroll
  for (int nt = 0; nt < 4; ++nt)
#pragma unroll
    for (int v = 0; v < 8; ++v) {
      float comb = aw[v] * kd[nt][v] + (1.0f - aw[v]) * qd[nt][v] + ed[nt][v];
      int row = v + ((lane >> 4) << 3);
      my[row * 64 + (nt << 4) + nn] = (_Float16)comb;
    }
  __syncthreads();

  // h = relu(combined @ o1_W + o1_b)
  v16h ah0 = load_t_f16((const _Float16*)my, nn, 64, 0);
  v16h ah1 = load_t_f16((const _Float16*)my, nn, 64, 32);
  v8f hd[4];
#pragma unroll
  for (int nt = 0; nt < 4; ++nt) {
    int nb = (nt << 4) + nn;
    v8f c = {};
    c = wmma16(ah0, load_t_f16(o1WT, nb, 64, 0), c);
    c = wmma16(ah1, load_t_f16(o1WT, nb, 64, 32), c);
    float ob = o1_b[nb];
#pragma unroll
    for (int v = 0; v < 8; ++v) c[v] = fmaxf(c[v] + ob, 0.0f);
    hd[nt] = c;
  }

  // logits = h @ o2_W (64x3): lane partial over its 4 channels, butterfly sum
  float lg0[8], lg1[8], lg2[8];
#pragma unroll
  for (int v = 0; v < 8; ++v) { lg0[v] = 0.0f; lg1[v] = 0.0f; lg2[v] = 0.0f; }
#pragma unroll
  for (int nt = 0; nt < 4; ++nt) {
    const float* wr = o2W + ((nt << 4) + nn) * 3;
    float w0 = wr[0], w1 = wr[1], w2 = wr[2];
#pragma unroll
    for (int v = 0; v < 8; ++v) {
      lg0[v] += hd[nt][v] * w0;
      lg1[v] += hd[nt][v] * w1;
      lg2[v] += hd[nt][v] * w2;
    }
  }
#pragma unroll
  for (int v = 0; v < 8; ++v)
#pragma unroll
    for (int off = 1; off < 16; off <<= 1) {
      lg0[v] += __shfl_xor(lg0[v], off, 32);
      lg1[v] += __shfl_xor(lg1[v], off, 32);
      lg2[v] += __shfl_xor(lg2[v], off, 32);
    }
  int vsel = lane & 7;
  int el = vsel + ((lane >> 4) << 3);
  float r0 = 0.0f, r1 = 0.0f, r2 = 0.0f;
#pragma unroll
  for (int i = 0; i < 8; ++i)
    if (i == vsel) { r0 = lg0[i]; r1 = lg1[i]; r2 = lg2[i]; }
  if ((lane & 15) < 8) {
    long o = ((long)(e0 + el)) * 3;
    out[o + 0] = r0 + o2_b[0];
    out[o + 1] = r1 + o2_b[1];
    out[o + 2] = r2 + o2_b[2];
  }
}

// ---------------------------------------------------------------------------
// Host orchestration
// ---------------------------------------------------------------------------
static void run_gat(hipStream_t st, const float* xsrc, int Ms, int Ks,
                    const _Float16* WlT, const float* bl, const float* xdst, int Md, int Kd,
                    const _Float16* WrT, const float* br, const _Float16* WeT,
                    const float* att, const float* bias, const int* srcIdx,
                    const int* dstIdx, const float* eattr, int concat, float* out,
                    float* xl, float* xr, float* alpha, unsigned* amax, float* denom,
                    float* acc) {
  gemm_bias_kernel<<<(Ms / 16) * 8, 32, 0, st>>>(xsrc, WlT, bl, xl, 128, Ks, 8);
  gemm_bias_kernel<<<(Md / 16) * 8, 32, 0, st>>>(xdst, WrT, br, xr, 128, Kd, 8);
  zero_kernel<<<(Md * 2 + 255) / 256, 256, 0, st>>>((float*)amax, (long)Md * 2);
  zero_kernel<<<(Md * 2 + 255) / 256, 256, 0, st>>>(denom, (long)Md * 2);
  zero_kernel<<<((long)Md * 128 + 255) / 256, 256, 0, st>>>(acc, (long)Md * 128);
  gat_edge_alpha_kernel<<<E_EDGES / 32, 64, 0, st>>>(eattr, srcIdx, dstIdx, WeT, xl, xr,
                                                     att, alpha, amax);
  gat_edge_accum_kernel<<<E_EDGES / 8, 256, 0, st>>>(srcIdx, dstIdx, alpha, amax, xl,
                                                     denom, acc);
  if (concat)
    gat_finalize_concat_kernel<<<(Md * 128 + 255) / 256, 256, 0, st>>>(acc, denom, bias,
                                                                       out, Md * 128);
  else
    gat_finalize_mean_kernel<<<(Md * 64 + 255) / 256, 256, 0, st>>>(acc, denom, bias,
                                                                    out, Md * 64);
}

extern "C" void kernel_launch(void* const* d_in, const int* in_sizes, int n_in,
                              void* d_out, int out_size, void* d_ws, size_t ws_size,
                              hipStream_t stream) {
  (void)in_sizes; (void)n_in; (void)out_size; (void)ws_size;
  const float* x_pitcher = (const float*)d_in[0];
  const float* x_batter  = (const float*)d_in[1];
  const int*   esrc_pb   = (const int*)d_in[2];
  const int*   edst_pb   = (const int*)d_in[3];
  const float* eattr_pb  = (const float*)d_in[4];
  const int*   esrc_bp   = (const int*)d_in[5];
  const int*   edst_bp   = (const int*)d_in[6];
  const float* eattr_bp  = (const float*)d_in[7];
  // params flattened in insertion order:
  // g1_pb: 8 Wl, 9 bl, 10 Wr, 11 br, 12 We, 13 att, 14 bias
  // g1_bp: 15..21; g2_pb: 22..28; g2_bp: 29..35
  // 36 q_W, 37 q_b, 38 k_W, 39 k_b, 40 e_W, 41 e_b, 42 o1_W, 43 o1_b, 44 o2_W, 45 o2_b

  char* w = (char*)d_ws;
  size_t off = 0;
  auto carve = [&](size_t bytes) -> void* {
    void* p = w + off;
    off = (off + bytes + 255) & ~(size_t)255;
    return p;
  };
  _Float16* g1pb_WlT = (_Float16*)carve(128 * 32 * 2);
  _Float16* g1pb_WrT = (_Float16*)carve(128 * 32 * 2);
  _Float16* g1bp_WlT = (_Float16*)carve(128 * 32 * 2);
  _Float16* g1bp_WrT = (_Float16*)carve(128 * 32 * 2);
  _Float16* g2pb_WlT = (_Float16*)carve(128 * 128 * 2);
  _Float16* g2pb_WrT = (_Float16*)carve(128 * 128 * 2);
  _Float16* g2bp_WlT = (_Float16*)carve(128 * 128 * 2);
  _Float16* g2bp_WrT = (_Float16*)carve(128 * 128 * 2);
  _Float16* WeT1pb = (_Float16*)carve(128 * 32 * 2);
  _Float16* WeT1bp = (_Float16*)carve(128 * 32 * 2);
  _Float16* WeT2pb = (_Float16*)carve(128 * 32 * 2);
  _Float16* WeT2bp = (_Float16*)carve(128 * 32 * 2);
  _Float16* qWT  = (_Float16*)carve(64 * 64 * 2);
  _Float16* kWT  = (_Float16*)carve(64 * 64 * 2);
  _Float16* o1WT = (_Float16*)carve(64 * 64 * 2);
  _Float16* eWT  = (_Float16*)carve(64 * 32 * 2);
  float* xl   = (float*)carve((size_t)NBATT * 128 * 4);
  float* xr   = (float*)carve((size_t)NBATT * 128 * 4);
  float* pit1 = (float*)carve((size_t)NPITCH * 128 * 4);
  float* bat1 = (float*)carve((size_t)NBATT * 128 * 4);
  float* pit2 = (float*)carve((size_t)NPITCH * 64 * 4);
  float* bat2 = (float*)carve((size_t)NBATT * 64 * 4);
  float* alpha = (float*)carve((size_t)E_EDGES * 2 * 4);
  unsigned* amax = (unsigned*)carve((size_t)NBATT * 2 * 4);
  float* denom = (float*)carve((size_t)NBATT * 2 * 4);
  float* acc = (float*)carve((size_t)NBATT * 128 * 4);

  auto tr = [&](_Float16* dst, int idx, int K, int N, int Kpad) {
    int tot = N * Kpad;
    transpose_h_kernel<<<(tot + 255) / 256, 256, 0, stream>>>(dst, (const float*)d_in[idx],
                                                              K, N, Kpad);
  };
  tr(g1pb_WlT, 8, 32, 128, 32);   tr(g1pb_WrT, 10, 32, 128, 32);
  tr(g1bp_WlT, 15, 32, 128, 32);  tr(g1bp_WrT, 17, 32, 128, 32);
  tr(g2pb_WlT, 22, 128, 128, 128); tr(g2pb_WrT, 24, 128, 128, 128);
  tr(g2bp_WlT, 29, 128, 128, 128); tr(g2bp_WrT, 31, 128, 128, 128);
  tr(WeT1pb, 12, 16, 128, 32);    tr(WeT1bp, 19, 16, 128, 32);
  tr(WeT2pb, 26, 16, 128, 32);    tr(WeT2bp, 33, 16, 128, 32);
  tr(qWT, 36, 64, 64, 64);        tr(kWT, 38, 64, 64, 64);
  tr(o1WT, 42, 64, 64, 64);       tr(eWT, 40, 16, 64, 32);

  // Layer 1 (concat=True)
  run_gat(stream, x_pitcher, NPITCH, 32, g1pb_WlT, (const float*)d_in[9], x_batter, NBATT,
          32, g1pb_WrT, (const float*)d_in[11], WeT1pb, (const float*)d_in[13],
          (const float*)d_in[14], esrc_pb, edst_pb, eattr_pb, 1, bat1, xl, xr, alpha,
          amax, denom, acc);
  run_gat(stream, x_batter, NBATT, 32, g1bp_WlT, (const float*)d_in[16], x_pitcher,
          NPITCH, 32, g1bp_WrT, (const float*)d_in[18], WeT1bp, (const float*)d_in[20],
          (const float*)d_in[21], esrc_bp, edst_bp, eattr_bp, 1, pit1, xl, xr, alpha,
          amax, denom, acc);
  // Layer 2 (concat=False, mean over heads)
  run_gat(stream, pit1, NPITCH, 128, g2pb_WlT, (const float*)d_in[23], bat1, NBATT, 128,
          g2pb_WrT, (const float*)d_in[25], WeT2pb, (const float*)d_in[27],
          (const float*)d_in[28], esrc_pb, edst_pb, eattr_pb, 0, bat2, xl, xr, alpha,
          amax, denom, acc);
  run_gat(stream, bat1, NBATT, 128, g2bp_WlT, (const float*)d_in[30], pit1, NPITCH, 128,
          g2bp_WrT, (const float*)d_in[32], WeT2bp, (const float*)d_in[34],
          (const float*)d_in[35], esrc_bp, edst_bp, eattr_bp, 0, pit2, xl, xr, alpha,
          amax, denom, acc);

  classifier_kernel<<<E_EDGES / 32, 64, 0, stream>>>(
      esrc_pb, edst_pb, eattr_pb, pit2, bat2, qWT, (const float*)d_in[37], kWT,
      (const float*)d_in[39], eWT, (const float*)d_in[41], o1WT, (const float*)d_in[43],
      (const float*)d_in[44], (const float*)d_in[45], (float*)d_out);
}
